// MultiheadAttention_23046794510346
// MI455X (gfx1250) — matrix operand order repaired and checked by reading, not compiled, not run on hardware
//
#include <hip/hip_runtime.h>
#include <hip/hip_bf16.h>
#include <stdint.h>

// ---- problem constants (match reference) ----
#define Bc     4
#define Sc     2048
#define Dc     1024
#define Hc     16
#define DKc    64
#define BSc    (Bc * Sc)          // 8192

typedef __attribute__((ext_vector_type(16))) _Float16 v16h;
typedef __attribute__((ext_vector_type(8)))  _Float16 v8h;
typedef __attribute__((ext_vector_type(8)))  float    v8f;
typedef __attribute__((ext_vector_type(4)))  unsigned int v4u;
typedef __attribute__((ext_vector_type(8)))  int      v8i;
typedef __attribute__((ext_vector_type(4)))  int      v4i;

static __device__ __forceinline__ v8f vzero8f() {
    v8f z = {0.f, 0.f, 0.f, 0.f, 0.f, 0.f, 0.f, 0.f};
    return z;
}

// D = A(16x32 f16) * B(32x16 f16) + C(16x16 f32)
static __device__ __forceinline__ v8f wmma32(v16h a, v16h b, v8f c) {
    return __builtin_amdgcn_wmma_f32_16x16x32_f16(
        /*neg_a=*/false, a, /*neg_b=*/false, b,
        /*c_mod=*/(short)0, c, /*reuse_a=*/false, /*reuse_b=*/false);
}

// A fragment, 16x32 (MxK), row-major source with leading dim lda (halves).
// Lanes 0-15: row=lane, K=[0..7],[16..23]; lanes 16-31: row=lane-16, K=[8..15],[24..31].
static __device__ __forceinline__ v16h load_a_frag(const _Float16* A, int lda,
                                                   int row0, int k0, int lane) {
    int r  = row0 + (lane & 15);
    int kh = (lane < 16) ? 0 : 8;
    const _Float16* p = A + (long)r * lda + k0 + kh;
    union { v16h v; v8h h[2]; } u;
    u.h[0] = *(const v8h*)(p);
    u.h[1] = *(const v8h*)(p + 16);
    return u.v;
}

// B fragment, 32x16 (KxN). Source "N-major": Bt[n][k] contiguous in k, leading dim ldb.
// Lanes 0-15: col=lane, K=[0..7],[16..23]; lanes 16-31: K=[8..15],[24..31].
static __device__ __forceinline__ v16h load_b_frag(const _Float16* Bt, int ldb,
                                                   int col0, int k0, int lane) {
    int c  = col0 + (lane & 15);
    int kh = (lane < 16) ? 0 : 8;
    const _Float16* p = Bt + (long)c * ldb + k0 + kh;
    union { v16h v; v8h h[2]; } u;
    u.h[0] = *(const v8h*)(p);
    u.h[1] = *(const v8h*)(p + 16);
    return u.v;
}

static __device__ __forceinline__ float rowmax16(float v) {
    v = fmaxf(v, __shfl_xor(v, 1, 32));
    v = fmaxf(v, __shfl_xor(v, 2, 32));
    v = fmaxf(v, __shfl_xor(v, 4, 32));
    v = fmaxf(v, __shfl_xor(v, 8, 32));
    return v;
}
static __device__ __forceinline__ float rowsum16(float v) {
    v += __shfl_xor(v, 1, 32);
    v += __shfl_xor(v, 2, 32);
    v += __shfl_xor(v, 4, 32);
    v += __shfl_xor(v, 8, 32);
    return v;
}

// ---- Tensor Data Mover: 2D tile (rows x cols of f16) -> LDS, row-major ----
// D# per CDNA5 ISA ch.8: group0 = {count, lds_addr, global_addr, type=2},
// group1 = {data_size=2B, tensor_dim0/1, tile_dim0/1, dim0_stride}. 2D tensor:
// groups 2/3 zero.
static __device__ __forceinline__ void tdm_load_tile_2d(
    unsigned int lds_off, const _Float16* gptr,
    unsigned int tensor_d0, unsigned int tensor_d1, unsigned int stride_d0,
    unsigned int tile_d0, unsigned int tile_d1)
{
    unsigned long long ga = (unsigned long long)(uintptr_t)gptr;
    v4u g0;
    g0[0] = 1u;                                        // count=1, user descriptor
    g0[1] = lds_off;                                   // LDS byte address
    g0[2] = (unsigned int)(ga & 0xffffffffu);          // global_addr[31:0]
    g0[3] = (unsigned int)((ga >> 32) & 0x01ffffffu)   // global_addr[56:32]
          | (2u << 30);                                // type=2 ("image")
    v8i g1;
    g1[0] = (int)(1u << 16);                           // data_size=1 (2 bytes)
    g1[1] = (int)((tensor_d0 & 0xffffu) << 16);        // tensor_dim0[15:0]
    g1[2] = (int)((tensor_d0 >> 16) | ((tensor_d1 & 0xffffu) << 16));
    g1[3] = (int)((tensor_d1 >> 16) | ((tile_d0 & 0xffffu) << 16));
    g1[4] = (int)(tile_d1 & 0xffffu);                  // tile_dim1 (tile_dim2=0)
    g1[5] = (int)stride_d0;                            // tensor_dim0_stride[31:0]
    g1[6] = 0;                                         // stride0[47:32], dim1_stride lo
    g1[7] = 0;
    v4i gz = {0, 0, 0, 0};
#if __clang_major__ >= 23
    v8i gz8 = {0, 0, 0, 0, 0, 0, 0, 0};
    __builtin_amdgcn_tensor_load_to_lds(g0, g1, gz, gz, gz8, 0);
#else
    __builtin_amdgcn_tensor_load_to_lds(g0, g1, gz, gz, 0);
#endif
}

// ---------------- stage 0: casts ----------------
__global__ void mha_cast_f16(const float* __restrict__ x,
                             _Float16* __restrict__ y, long n) {
    long i = (long)blockIdx.x * blockDim.x + threadIdx.x;
    if (i < n) y[i] = (_Float16)x[i];
}

// W: [H][D][DK] f32  ->  WT: [H][DK][D] f16
__global__ void mha_transpose_w(const float* __restrict__ w,
                                _Float16* __restrict__ wt) {
    long i = (long)blockIdx.x * blockDim.x + threadIdx.x;
    long total = (long)Hc * Dc * DKc;
    if (i >= total) return;
    int kk = (int)(i % DKc);
    long t = i / DKc;
    int d  = (int)(t % Dc);
    int h  = (int)(t / Dc);
    wt[((long)h * DKc + kk) * Dc + d] = (_Float16)w[i];
}

// ---------------- stage 1: per-head projection GEMM ----------------
__global__ __launch_bounds__(256)
void mha_proj_gemm(const _Float16* __restrict__ X,   // [BS][D] f16
                   const _Float16* __restrict__ WT,  // [H][DK][D] f16
                   const float*    __restrict__ bias,// [H][DK] f32
                   _Float16* __restrict__ out, int transposeOut) {
    int lane = threadIdx.x & 31;
    int wave = threadIdx.x >> 5;
    int h    = blockIdx.z;
    long m0  = ((long)blockIdx.x * 8 + wave) * 64;
    const _Float16* A  = X + m0 * Dc;
    const _Float16* Bt = WT + (long)h * DKc * Dc;

    v8f acc[4][4];
    for (int mt = 0; mt < 4; ++mt)
        for (int nt = 0; nt < 4; ++nt) acc[mt][nt] = vzero8f();

    for (int k0 = 0; k0 < Dc; k0 += 32) {
        v16h af[4], bf[4];
        for (int mt = 0; mt < 4; ++mt) af[mt] = load_a_frag(A, Dc, mt * 16, k0, lane);
        for (int nt = 0; nt < 4; ++nt) bf[nt] = load_b_frag(Bt, Dc, nt * 16, k0, lane);
        for (int mt = 0; mt < 4; ++mt)
            for (int nt = 0; nt < 4; ++nt)
                acc[mt][nt] = wmma32(af[mt], bf[nt], acc[mt][nt]);
    }

    int rowoff = (lane < 16) ? 0 : 8;
    int c      = lane & 15;
    for (int mt = 0; mt < 4; ++mt) {
        for (int r = 0; r < 8; ++r) {
            long m = m0 + mt * 16 + r + rowoff;
            int b  = (int)(m / Sc);
            int s  = (int)(m % Sc);
            for (int nt = 0; nt < 4; ++nt) {
                int n = nt * 16 + c;
                float v = acc[mt][nt][r] + bias[h * DKc + n];
                long idx = transposeOut
                    ? ((((long)b * Hc + h) * DKc + n) * Sc + s)   // [b,h,dk,s]
                    : ((((long)b * Hc + h) * Sc + s) * DKc + n);  // [b,h,s,dk]
                out[idx] = (_Float16)v;
            }
        }
    }
}

// ---------------- stage 2: fused flash attention ----------------
// One wave handles 32 query rows; all 8 waves of a block share one (b,h) and
// sweep the same K/V tiles, which are staged into LDS once per block by the
// Tensor Data Mover. grid = 512 blocks of 256.
__global__ __launch_bounds__(256)
void mha_flash_attn(const _Float16* __restrict__ qp,  // [b,h,s,dk] f16
                    const _Float16* __restrict__ kp,  // [b,h,s,dk] f16
                    const _Float16* __restrict__ vT,  // [b,h,dk,s] f16
                    _Float16* __restrict__ concat) {  // [b,s,dk,h] f16
    __shared__ _Float16 Pbuf[8][32][72];   // per-wave P relayout tile (36 KB)
    __shared__ _Float16 Ktile[64 * 64];    // [key][d]   (8 KB, TDM-filled)
    __shared__ _Float16 Vtile[64 * 64];    // [dk][s]    (8 KB, TDM-filled)

    int lane = threadIdx.x & 31;
    int wave = threadIdx.x >> 5;
    int tile = blockIdx.x * 8 + wave;      // 32-row q tile id, 0..4095
    int bh   = tile >> 6;                  // same for all 8 waves of a block
    int qt   = tile & 63;

    const _Float16* qh = qp + (long)bh * Sc * DKc + (long)qt * 32 * DKc;
    const _Float16* kh = kp + (long)bh * Sc * DKc;
    const _Float16* vh = vT + (long)bh * DKc * Sc;

    // Preload the 32x64 Q tile as A fragments, folding in 1/sqrt(d_k) = 1/8.
    v16h qa[2][2];
    const _Float16 hscale = (_Float16)0.125f;
    for (int mt = 0; mt < 2; ++mt)
        for (int ks = 0; ks < 2; ++ks) {
            v16h a = load_a_frag(qh, DKc, mt * 16, ks * 32, lane);
            for (int i = 0; i < 16; ++i) a[i] = a[i] * hscale;
            qa[mt][ks] = a;
        }

    v8f acc[2][4];
    float mstate[2][8], lstate[2][8];
    for (int mt = 0; mt < 2; ++mt) {
        for (int nt = 0; nt < 4; ++nt) acc[mt][nt] = vzero8f();
        for (int r = 0; r < 8; ++r) { mstate[mt][r] = -3.0e38f; lstate[mt][r] = 0.f; }
    }

    int rowoff = (lane < 16) ? 0 : 8;
    int c      = lane & 15;

    for (int kt = 0; kt < Sc / 64; ++kt) {
        __syncthreads();                   // tiles from previous iter fully consumed
        if (wave == 0) {
            // K tile: rows kt*64..+63 of [S][DK] -> Ktile[key][d]
            tdm_load_tile_2d((unsigned int)(uintptr_t)Ktile,
                             kh + (long)kt * 64 * DKc,
                             /*tensor_d0=*/DKc, /*tensor_d1=*/Sc,
                             /*stride_d0=*/DKc, /*tile_d0=*/64, /*tile_d1=*/64);
            // V tile: cols kt*64..+63 of [DK][S] -> Vtile[dk][s]
            tdm_load_tile_2d((unsigned int)(uintptr_t)Vtile,
                             vh + (long)kt * 64,
                             /*tensor_d0=*/Sc, /*tensor_d1=*/DKc,
                             /*stride_d0=*/Sc, /*tile_d0=*/64, /*tile_d1=*/64);
            __builtin_amdgcn_s_wait_tensorcnt(0);
        }
        __syncthreads();                   // publish TDM-filled tiles

        // scores = (Q/8) @ K^T for this block of 64 keys
        v8f sc[2][4];
        for (int mt = 0; mt < 2; ++mt)
            for (int nt = 0; nt < 4; ++nt) sc[mt][nt] = vzero8f();
        for (int ks = 0; ks < 2; ++ks) {
            v16h bk[4];
            for (int nt = 0; nt < 4; ++nt)
                bk[nt] = load_b_frag(Ktile, 64, nt * 16, ks * 32, lane);
            for (int mt = 0; mt < 2; ++mt)
                for (int nt = 0; nt < 4; ++nt)
                    sc[mt][nt] = wmma32(qa[mt][ks], bk[nt], sc[mt][nt]);
        }

        // online softmax (per row)
        for (int mt = 0; mt < 2; ++mt) {
            for (int r = 0; r < 8; ++r) {
                float pm = fmaxf(fmaxf(sc[mt][0][r], sc[mt][1][r]),
                                 fmaxf(sc[mt][2][r], sc[mt][3][r]));
                pm = rowmax16(pm);
                float mold = mstate[mt][r];
                float mnew = fmaxf(mold, pm);
                float corr = __expf(mold - mnew);
                mstate[mt][r] = mnew;
                float psum = 0.f;
                for (int nt = 0; nt < 4; ++nt) {
                    float e = __expf(sc[mt][nt][r] - mnew);
                    sc[mt][nt][r] = e;
                    psum += e;
                }
                psum = rowsum16(psum);
                lstate[mt][r] = lstate[mt][r] * corr + psum;
                for (int nt = 0; nt < 4; ++nt) acc[mt][nt][r] *= corr;
            }
        }

        // C-layout -> A-layout relayout of P through per-wave LDS tile
        for (int mt = 0; mt < 2; ++mt)
            for (int nt = 0; nt < 4; ++nt)
                for (int r = 0; r < 8; ++r)
                    Pbuf[wave][mt * 16 + r + rowoff][nt * 16 + c] =
                        (_Float16)sc[mt][nt][r];

        // out += P @ V (V tile transposed in LDS: contiguous-K B fragments)
        for (int ks = 0; ks < 2; ++ks) {
            v16h pa[2];
            for (int mt = 0; mt < 2; ++mt)
                pa[mt] = load_a_frag(&Pbuf[wave][0][0], 72, mt * 16, ks * 32, lane);
            v16h bv[4];
            for (int nt = 0; nt < 4; ++nt)
                bv[nt] = load_b_frag(Vtile, 64, nt * 16, ks * 32, lane);
            for (int mt = 0; mt < 2; ++mt)
                for (int nt = 0; nt < 4; ++nt)
                    acc[mt][nt] = wmma32(pa[mt], bv[nt], acc[mt][nt]);
        }
    }

    // epilogue: normalize and scatter into concat layout (b, s, dk, h)
    int b = bh >> 4;
    int h = bh & 15;
    for (int mt = 0; mt < 2; ++mt) {
        for (int r = 0; r < 8; ++r) {
            float inv = 1.0f / lstate[mt][r];
            int s = qt * 32 + mt * 16 + r + rowoff;
            for (int nt = 0; nt < 4; ++nt) {
                int dk = nt * 16 + c;
                concat[(((long)b * Sc + s) * DKc + dk) * Hc + h] =
                    (_Float16)(acc[mt][nt][r] * inv);
            }
        }
    }
}

// ---------------- stage 3: output projection ----------------
__global__ __launch_bounds__(256)
void mha_out_gemm(const _Float16* __restrict__ A,   // [BS][D] f16
                  const _Float16* __restrict__ Woh, // [D][D] f16 (row n contiguous in d)
                  const float*    __restrict__ bo,  // [D] f32
                  float* __restrict__ out) {        // [BS][D] f32
    int lane = threadIdx.x & 31;
    int wave = threadIdx.x >> 5;
    long m0  = ((long)blockIdx.x * 8 + wave) * 64;
    int n0   = blockIdx.y * 64;
    const _Float16* Am = A + m0 * Dc;
    const _Float16* Bt = Woh + (long)n0 * Dc;

    v8f acc[4][4];
    for (int mt = 0; mt < 4; ++mt)
        for (int nt = 0; nt < 4; ++nt) acc[mt][nt] = vzero8f();

    for (int k0 = 0; k0 < Dc; k0 += 32) {
        v16h af[4], bf[4];
        for (int mt = 0; mt < 4; ++mt) af[mt] = load_a_frag(Am, Dc, mt * 16, k0, lane);
        for (int nt = 0; nt < 4; ++nt) bf[nt] = load_b_frag(Bt, Dc, nt * 16, k0, lane);
        for (int mt = 0; mt < 4; ++mt)
            for (int nt = 0; nt < 4; ++nt)
                acc[mt][nt] = wmma32(af[mt], bf[nt], acc[mt][nt]);
    }

    int rowoff = (lane < 16) ? 0 : 8;
    int c      = lane & 15;
    for (int mt = 0; mt < 4; ++mt)
        for (int r = 0; r < 8; ++r) {
            long m = m0 + mt * 16 + r + rowoff;
            for (int nt = 0; nt < 4; ++nt) {
                int n = n0 + nt * 16 + c;
                out[m * Dc + n] = acc[mt][nt][r] + bo[n];
            }
        }
}

// ---------------- host launcher ----------------
extern "C" void kernel_launch(void* const* d_in, const int* in_sizes, int n_in,
                              void* d_out, int out_size, void* d_ws, size_t ws_size,
                              hipStream_t stream) {
    (void)in_sizes; (void)n_in; (void)out_size; (void)ws_size;
    // setup_inputs order: K, Q, V, Wq, bq, Wk, bk, Wv, bv, Wo, bo
    const float* Kf  = (const float*)d_in[0];
    const float* Qf  = (const float*)d_in[1];
    const float* Vf  = (const float*)d_in[2];
    const float* Wq  = (const float*)d_in[3];
    const float* bq  = (const float*)d_in[4];
    const float* Wk  = (const float*)d_in[5];
    const float* bk  = (const float*)d_in[6];
    const float* Wv  = (const float*)d_in[7];
    const float* bv  = (const float*)d_in[8];
    const float* Wo  = (const float*)d_in[9];
    const float* bo  = (const float*)d_in[10];
    float* out = (float*)d_out;

    size_t off = 0;
    auto alloc = [&](size_t bytes) -> void* {
        void* p = (char*)d_ws + off;
        off += (bytes + 255) & ~(size_t)255;
        return p;
    };
    const long nQKV = (long)BSc * Dc;           // 8388608
    const long nW   = (long)Hc * Dc * DKc;      // 1048576
    const long nPRJ = (long)Bc * Hc * Sc * DKc; // 8388608

    _Float16* Qh    = (_Float16*)alloc(nQKV * 2);
    _Float16* Kh    = (_Float16*)alloc(nQKV * 2);
    _Float16* Vh    = (_Float16*)alloc(nQKV * 2);
    _Float16* WqT   = (_Float16*)alloc(nW * 2);
    _Float16* WkT   = (_Float16*)alloc(nW * 2);
    _Float16* WvT   = (_Float16*)alloc(nW * 2);
    _Float16* Woh   = (_Float16*)alloc((long)Dc * Dc * 2);
    _Float16* qprj  = (_Float16*)alloc(nPRJ * 2);
    _Float16* kprj  = (_Float16*)alloc(nPRJ * 2);
    _Float16* vTprj = (_Float16*)alloc(nPRJ * 2);
    _Float16* conc  = (_Float16*)alloc(nQKV * 2);

    const int T = 256;
    // stage 0: casts
    mha_cast_f16<<<(int)((nQKV + T - 1) / T), T, 0, stream>>>(Qf, Qh, nQKV);
    mha_cast_f16<<<(int)((nQKV + T - 1) / T), T, 0, stream>>>(Kf, Kh, nQKV);
    mha_cast_f16<<<(int)((nQKV + T - 1) / T), T, 0, stream>>>(Vf, Vh, nQKV);
    mha_cast_f16<<<(int)(((long)Dc * Dc + T - 1) / T), T, 0, stream>>>(Wo, Woh, (long)Dc * Dc);
    mha_transpose_w<<<(int)((nW + T - 1) / T), T, 0, stream>>>(Wq, WqT);
    mha_transpose_w<<<(int)((nW + T - 1) / T), T, 0, stream>>>(Wk, WkT);
    mha_transpose_w<<<(int)((nW + T - 1) / T), T, 0, stream>>>(Wv, WvT);

    // stage 1: projections (8192x64 per head, K=1024)
    dim3 pgrid(BSc / (64 * 8), 1, Hc);
    mha_proj_gemm<<<pgrid, T, 0, stream>>>(Qh, WqT, bq, qprj, 0);
    mha_proj_gemm<<<pgrid, T, 0, stream>>>(Kh, WkT, bk, kprj, 0);
    mha_proj_gemm<<<pgrid, T, 0, stream>>>(Vh, WvT, bv, vTprj, 1);

    // stage 2: flash attention (4096 waves, 8 per block, TDM-staged K/V tiles)
    mha_flash_attn<<<512, T, 0, stream>>>(qprj, kprj, vTprj, conc);

    // stage 3: output projection (8192x1024, K=1024)
    dim3 ogrid(BSc / (64 * 8), Dc / 64, 1);
    mha_out_gemm<<<ogrid, T, 0, stream>>>(conc, Woh, bo, out);
}
